// BSG_78829829751266
// MI455X (gfx1250) — matrix operand order repaired and checked by reading, not compiled, not run on hardware
//
#include <hip/hip_runtime.h>

typedef __attribute__((ext_vector_type(16))) __bf16 v16bf;
typedef __attribute__((ext_vector_type(8)))  float  v8f;

constexpr int Dm = 50;    // input dim
constexpr int Hh = 50;    // hidden dim
constexpr int Ll = 100;   // latent dim
constexpr int Cc = 10;    // 2*window
constexpr int TB = 16;    // batch rows per block
constexpr int NW = 10;    // waves per block (one GEMM1 M-tile each)
constexpr int NT = NW * 32;

union FragB { v16bf v; uint4 q[2]; };
union PkBf  { unsigned u; __bf16 b[2]; };

__global__ void bsg_zero_out(float* out) { out[0] = 0.0f; }

__global__ __launch_bounds__(NT) void bsg_fused(
    const float* __restrict__ emb,  const float* __restrict__ W1,
    const float* __restrict__ b1,   const float* __restrict__ Wmu,
    const float* __restrict__ bmu,  const float* __restrict__ Wls,
    const float* __restrict__ bls,  const float* __restrict__ tmeans,
    const float* __restrict__ tlogv,const int* __restrict__ centers,
    const int* __restrict__ contexts, const int* __restrict__ negctx,
    float* __restrict__ out, float invB)
{
  extern __shared__ __align__(16) char smem[];
  // bf16 staging (all 16B aligned)
  __bf16* W1T  = (__bf16*)smem;          // [64 N][128 K]  = 16384 B
  __bf16* WmuT = W1T  + 64 * 128;        // [112 N][64 K]  = 14336 B
  __bf16* catA = WmuT + 112 * 64;        // [160 row][128] = 40960 B
  __bf16* hbf  = catA + 160 * 128;       // [16 row][64]   =  2048 B
  // f32 working set
  float* h     = (float*)(hbf + TB * 64);// [16][64]
  float* mu    = h    + TB * 64;         // [16][112]
  float* lsv   = mu   + TB * 112;        // [16]
  float* b1s   = lsv  + TB;              // [64]
  float* bmus  = b1s  + 64;              // [112]
  float* wlss  = bmus + 112;             // [64]
  float* klc   = wlss + 64;              // [16][10]
  float* kln   = klc  + TB * Cc;         // [16][10]
  float* klcen = kln  + TB * Cc;         // [16]
  float* redacc= klcen+ TB;              // [1]

  const int t    = threadIdx.x;
  const int wave = t >> 5;
  const int lane = t & 31;
  const int hl   = lane >> 4;            // lane half (0/1)
  const int lrow = lane & 15;            // M (A) or N (B, C/D) index within tile
  const int b0   = blockIdx.x * TB;

  // ---- Phase 0: stage weights (transposed, bf16, zero-padded) + init ----
  for (int i = t; i < 64 * 128; i += NT) {
    int n = i >> 7, k = i & 127;
    W1T[i] = (__bf16)((n < Hh && k < 2 * Dm) ? W1[k * Hh + n] : 0.0f);
  }
  for (int i = t; i < 112 * 64; i += NT) {
    int n = i >> 6, k = i & 63;
    WmuT[i] = (__bf16)((n < Ll && k < Hh) ? Wmu[k * Ll + n] : 0.0f);
  }
  if (t < 64)  { b1s[t] = (t < Hh) ? b1[t] : 0.0f; wlss[t] = (t < Hh) ? Wls[t] : 0.0f; }
  if (t < 112) { bmus[t] = (t < Ll) ? bmu[t] : 0.0f; }
  for (int i = t; i < TB * 64; i += NT) h[i] = 0.0f;
  if (t == 0) redacc[0] = 0.0f;

  // ---- Phase 1: gather + stage cat rows [160 x 128] bf16 (2 threads/row) ----
  // Pairwise: Dm is even, emb rows are 8B-aligned -> float2 loads never straddle
  // the ctx/center boundary; packed bf16 pairs stored as b32.
  {
    int r = t >> 1, hh = t & 1;                 // 160 rows x 2 halves = 320 tasks
    int b = r / Cc, c = r % Cc;
    int ctx = contexts[(b0 + b) * Cc + c];
    int cen = centers[b0 + b];
    const float* e0 = emb + (size_t)ctx * Dm;
    const float* e1 = emb + (size_t)cen * Dm;
    __bf16* dst = catA + r * 128 + hh * 64;
    #pragma unroll
    for (int kp = 0; kp < 32; ++kp) {
      int kk = hh * 64 + kp * 2;
      float2 v;
      if (kk < 2 * Dm) {
        const float* src = (kk < Dm) ? (e0 + kk) : (e1 + (kk - Dm));
        v = *(const float2*)src;
      } else {
        v.x = 0.0f; v.y = 0.0f;
      }
      PkBf pk;
      pk.b[0] = (__bf16)v.x;
      pk.b[1] = (__bf16)v.y;
      *(unsigned*)(dst + kp * 2) = pk.u;
    }
  }
  __syncthreads();

  // ---- Phase 2: GEMM1  (16x64 tile per wave, K=128 in 4 steps) ----
  {
    // A fragments: lane half hl: VGPRs0-3 <- K = ks*32 + hl*8 .. +7 ; VGPRs4-7 <- +16
    FragB a[4];
    #pragma unroll
    for (int ks = 0; ks < 4; ++ks) {
      const __bf16* p = catA + (wave * 16 + lrow) * 128 + ks * 32 + hl * 8;
      a[ks].q[0] = *(const uint4*)p;
      a[ks].q[1] = *(const uint4*)(p + 16);
    }
    v8f z = {0.f,0.f,0.f,0.f,0.f,0.f,0.f,0.f};
    v8f acc[4] = {z, z, z, z};
    #pragma unroll
    for (int nt = 0; nt < 4; ++nt) {
      #pragma unroll
      for (int ks = 0; ks < 4; ++ks) {
        // B fragment: lane = N; K = ks*32 + hl*8 .. +7 ; then +16
        FragB bb;
        const __bf16* p = W1T + (nt * 16 + lrow) * 128 + ks * 32 + hl * 8;
        bb.q[0] = *(const uint4*)p;
        bb.q[1] = *(const uint4*)(p + 16);
        acc[nt] = __builtin_amdgcn_wmma_f32_16x16x32_bf16(
            false, a[ks].v, false, bb.v, (short)0, acc[nt], false, false);
      }
    }
    // bias + ReLU, run-length sum over consecutive rows of the same batch
    // element (<=2 distinct b per lane), then LDS atomic: h[b][N] += sum
    #pragma unroll
    for (int nt = 0; nt < 4; ++nt) {
      int N = nt * 16 + lrow;
      if (N < Hh) {
        int   curb = (wave * 16 + hl * 8) / Cc;
        float run  = 0.0f;
        #pragma unroll
        for (int g = 0; g < 8; ++g) {
          int row = wave * 16 + hl * 8 + g;     // C/D layout: M = g + 8*half
          int b = row / Cc;
          float v = acc[nt][g] + b1s[N];
          v = v > 0.0f ? v : 0.0f;
          if (b != curb) {
            atomicAdd(&h[curb * 64 + N], run);
            run = 0.0f;
            curb = b;
          }
          run += v;
        }
        atomicAdd(&h[curb * 64 + N], run);
      }
    }
  }
  __syncthreads();

  // ---- Phase 3: h -> bf16 for GEMM2 ----
  for (int i = t; i < TB * 64; i += NT) hbf[i] = (__bf16)h[i];
  __syncthreads();

  // ---- Phase 4: GEMM2 (mu = h @ Wmu), waves 0..6 cover N=0..111; wave 7: logsigma ----
  if (wave < 7) {
    v8f z = {0.f,0.f,0.f,0.f,0.f,0.f,0.f,0.f};
    v8f acc = z;
    #pragma unroll
    for (int ks = 0; ks < 2; ++ks) {
      FragB a, bb;
      const __bf16* pa = hbf + lrow * 64 + ks * 32 + hl * 8;
      a.q[0] = *(const uint4*)pa;
      a.q[1] = *(const uint4*)(pa + 16);
      const __bf16* pb = WmuT + (wave * 16 + lrow) * 64 + ks * 32 + hl * 8;
      bb.q[0] = *(const uint4*)pb;
      bb.q[1] = *(const uint4*)(pb + 16);
      acc = __builtin_amdgcn_wmma_f32_16x16x32_bf16(
          false, a.v, false, bb.v, (short)0, acc, false, false);
    }
    int N = wave * 16 + lrow;
    #pragma unroll
    for (int g = 0; g < 8; ++g) {
      int M = g + hl * 8;                       // batch row in tile
      mu[M * 112 + N] = acc[g] + bmus[N];
    }
  } else if (wave == 7 && lane < TB) {
    float s = bls[0];
    for (int k = 0; k < Hh; ++k) s += h[lane * 64 + k] * wlss[k];
    lsv[lane] = s;
  }
  __syncthreads();

  // ---- Phase 5: KL (21 items per sample; one item per wave iteration),
  //      software-pipelined with global_prefetch of the next mean row ----
  {
    auto fetch_idx = [&](int it) -> int {
      int bb = it / 21, jj = it % 21;
      if (jj < Cc)       return contexts[(b0 + bb) * Cc + jj];
      if (jj < 2 * Cc)   return negctx[(b0 + bb) * Cc + (jj - Cc)];
      return centers[b0 + bb];
    };
    int item = wave;
    int idx  = (item < TB * 21) ? fetch_idx(item) : 0;
    while (item < TB * 21) {
      int nitem = item + NW;
      int nidx  = (nitem < TB * 21) ? fetch_idx(nitem) : 0;
      if (nitem < TB * 21 && lane < 4) {
        // pull next 400B mean row toward the WGP while we reduce this one
        __builtin_prefetch(tmeans + (size_t)nidx * Ll + lane * 32, 0, 1);
      }
      int b = item / 21, j = item % 21;
      const float* tm = tmeans + (size_t)idx * Ll;
      float ds = 0.0f;
      for (int k = lane; k < Ll; k += 32) {
        float d = mu[b * 112 + k] - tm[k];
        ds += d * d;
      }
      #pragma unroll
      for (int off = 16; off > 0; off >>= 1) ds += __shfl_xor(ds, off, 32);
      if (lane == 0) {
        float tl  = tlogv[idx];
        float tv  = __expf(tl);
        float lsb = lsv[b];
        float kl = 0.5f * (__expf(lsb) / tv + ds / tv + (tl - lsb) - (float)Ll);
        if (j < Cc)          klc[b * Cc + j] = kl;
        else if (j < 2 * Cc) kln[b * Cc + (j - Cc)] = kl;
        else                 klcen[b] = kl;
      }
      item = nitem;
      idx  = nidx;
    }
  }
  __syncthreads();

  // ---- Phase 6: hinge + block reduction + global atomic (pre-scaled by 1/B) ----
  {
    float p = 0.0f;
    bool has = false;
    if (t < TB * Cc) {
      float hv = klc[t] - kln[t] + 1.0f;        // MARGIN = 1.0
      p = hv > 0.0f ? hv : 0.0f;
      has = true;
    } else if (t < TB * Cc + TB) {
      p = klcen[t - TB * Cc];
      has = true;
    }
    if (has) atomicAdd(&redacc[0], p);
    __syncthreads();
    if (t == 0) atomicAdd(out, redacc[0] * invB);
  }
}

extern "C" void kernel_launch(void* const* d_in, const int* in_sizes, int n_in,
                              void* d_out, int out_size, void* d_ws, size_t ws_size,
                              hipStream_t stream) {
  const float* emb   = (const float*)d_in[0];
  const float* W1    = (const float*)d_in[1];
  const float* b1    = (const float*)d_in[2];
  const float* Wmu   = (const float*)d_in[3];
  const float* bmu   = (const float*)d_in[4];
  const float* Wls   = (const float*)d_in[5];
  const float* bls   = (const float*)d_in[6];
  const float* tm    = (const float*)d_in[7];
  const float* tl    = (const float*)d_in[8];
  const int* centers = (const int*)d_in[9];
  const int* ctx     = (const int*)d_in[10];
  const int* neg     = (const int*)d_in[11];
  float* out = (float*)d_out;

  int B = in_sizes[9];
  bsg_zero_out<<<1, 1, 0, stream>>>(out);

  int blocks = (B + TB - 1) / TB;
  // bf16: (8192 + 7168 + 20480 + 1024)*2 = 73728 B ; f32 tail ~13.7 KB
  size_t shmem = 87424;
  bsg_fused<<<blocks, NT, shmem, stream>>>(emb, W1, b1, Wmu, bmu, Wls, bls,
                                           tm, tl, centers, ctx, neg, out,
                                           1.0f / (float)B);
}
